// Model_59030030516745
// MI455X (gfx1250) — matrix operand order repaired and checked by reading, not compiled, not run on hardware
//
#include <hip/hip_runtime.h>
#include <hip/hip_bf16.h>

// ---------------- problem dims ----------------
constexpr int B_   = 32;
constexpr int L_   = 512;
constexpr int CIN_ = 16;
constexpr int D_   = 512;
constexpr int H_   = 8;
constexpr int DFF_ = 2048;
constexpr int NL_  = 3;
constexpr int NC_  = 10;
constexpr int PH_  = 128;
constexpr int DH_  = 64;
constexpr int M_   = B_ * L_;   // 16384 token rows

// ---------------- types ----------------
typedef __bf16 bf16_t;
typedef __attribute__((ext_vector_type(8)))  bf16_t v8bf;
typedef __attribute__((ext_vector_type(16))) bf16_t v16bf;
typedef __attribute__((ext_vector_type(8)))  float  v8f;
typedef unsigned int tdm_v4u __attribute__((ext_vector_type(4)));
typedef int          tdm_v8i __attribute__((ext_vector_type(8)));
typedef int          tdm_v4i __attribute__((ext_vector_type(4)));

__device__ __forceinline__ float gelu_exact(float x) {
    return 0.5f * x * (1.0f + erff(x * 0.7071067811865476f));
}

// =====================================================================
// Tensor Data Mover: 2D tile (tile_rows x tile_k bf16) global -> LDS.
// LDS rows are padded by TDM: pad_interval = 128B (32 dwords, code 4),
// pad_amount = 16B (4 dwords, code 3)  => LDS row stride = 144 bytes.
// =====================================================================
constexpr int ROWB = 144;    // LDS bytes per (64 bf16) tile row incl. TDM pad

__device__ __forceinline__ void tdm_load_2d(unsigned lds_addr, const bf16_t* gptr,
                                            unsigned tile_k, unsigned tile_rows,
                                            unsigned row_stride_elems)
{
    unsigned long long ga = (unsigned long long)(uintptr_t)gptr;
    tdm_v4u g0;
    g0.x = 1u;                                           // count=1 (valid user desc)
    g0.y = lds_addr;                                     // LDS byte address
    g0.z = (unsigned)(ga & 0xffffffffu);                 // global_addr[31:0]
    g0.w = (unsigned)((ga >> 32) & 0x01ffffffu) | (2u << 30);  // addr[56:32] | type=2
    tdm_v8i g1;
    g1[0] = (int)((1u << 16)     // data_size = 1 -> 2 bytes
                | (1u << 20)     // pad_enable
                | (4u << 22)     // pad_interval code 4 = 32 dwords (128B)
                | (3u << 25));   // pad_amount  code 3 = 4 dwords (16B)
    g1[1] = (int)((tile_k & 0xffffu) << 16);             // tensor_dim0[15:0]
    g1[2] = (int)((tile_k >> 16) | ((tile_rows & 0xffffu) << 16)); // d0 hi | tensor_dim1 lo
    g1[3] = (int)((tile_rows >> 16) | ((tile_k & 0xffffu) << 16)); // d1 hi | tile_dim0
    g1[4] = (int)(tile_rows & 0xffffu);                  // tile_dim1 | tile_dim2=0
    g1[5] = (int)row_stride_elems;                       // tensor_dim0_stride[31:0]
    g1[6] = 0;                                           // stride hi | dim1_stride lo
    g1[7] = 0;
    tdm_v4i gz = {0, 0, 0, 0};
#if defined(__clang_major__) && (__clang_major__ >= 23)
    tdm_v8i gz8 = {0, 0, 0, 0, 0, 0, 0, 0};
    __builtin_amdgcn_tensor_load_to_lds(g0, g1, gz, gz, gz8, 0);
#else
    __builtin_amdgcn_tensor_load_to_lds(g0, g1, gz, gz, 0);
#endif
}

// =====================================================================
// bf16 WMMA GEMM with TDM double-buffered staging:
//   C[b] = act(A[b] @ B[b]^T + bias)
// A: M x K bf16 (row-major, lda), B: N x K bf16 (row-major, ldb).
// Tile: BM=128, BN=BN_T (128 or 64), K-stage=64. 256 threads = 8 waves,
// wave grid 4x2; each wave computes 32 x (BN_T/2) via 16x16x32 WMMA.
// Requires M%128==0, N%BN_T==0, K%64==0 (true for all uses here).
// ACT: 0=none, 2=exact GELU.  OUT_BF16: 1 -> store bf16, 0 -> store f32.
// =====================================================================
constexpr int BM  = 128;
constexpr int BKS = 64;     // K elements per TDM stage

template<int BN_T, int ACT, int OUT_BF16>
__global__ __launch_bounds__(256)
void gemm_bf16_tdm(const bf16_t* __restrict__ A, int lda, long strideA,
                   const bf16_t* __restrict__ Bw, int ldb, long strideB,
                   void* __restrict__ Cv, int ldc, long strideC,
                   const float* __restrict__ bias, int K)
{
    constexpr int NJ = BN_T / 32;                 // 16-col tiles per wave
    __shared__ alignas(16) char Asm[2][BM   * ROWB];
    __shared__ alignas(16) char Bsm[2][BN_T * ROWB];

    const int tid  = threadIdx.x;
    const int wid  = tid >> 5;
    const int lane = tid & 31;
    const int wm   = wid >> 1;       // 0..3
    const int wn   = wid & 1;        // 0..1
    const int half = lane >> 4;
    const int lrow = lane & 15;

    const bf16_t* Ab = A  + (long)blockIdx.z * strideA + (long)blockIdx.y * BM   * lda;
    const bf16_t* Bb = Bw + (long)blockIdx.z * strideB + (long)blockIdx.x * BN_T * ldb;

    const v8f vzero = {0.f, 0.f, 0.f, 0.f, 0.f, 0.f, 0.f, 0.f};
    v8f acc[2][NJ];
    #pragma unroll
    for (int i = 0; i < 2; ++i)
        #pragma unroll
        for (int j = 0; j < NJ; ++j) acc[i][j] = vzero;

    const unsigned ldsA0 = (unsigned)(uintptr_t)&Asm[0][0];
    const unsigned ldsA1 = (unsigned)(uintptr_t)&Asm[1][0];
    const unsigned ldsB0 = (unsigned)(uintptr_t)&Bsm[0][0];
    const unsigned ldsB1 = (unsigned)(uintptr_t)&Bsm[1][0];

    const int nk = K / BKS;
    if (wid == 0) {
        tdm_load_2d(ldsA0, Ab, BKS, BM,   (unsigned)lda);
        tdm_load_2d(ldsB0, Bb, BKS, BN_T, (unsigned)ldb);
        __builtin_amdgcn_s_wait_tensorcnt(0);
    }
    __syncthreads();

    for (int kk = 0; kk < nk; ++kk) {
        const int cur = kk & 1;
        if (wid == 0 && (kk + 1) < nk) {
            tdm_load_2d(cur ? ldsA0 : ldsA1, Ab + (kk + 1) * BKS, BKS, BM,   (unsigned)lda);
            tdm_load_2d(cur ? ldsB0 : ldsB1, Bb + (kk + 1) * BKS, BKS, BN_T, (unsigned)ldb);
        }
        const char* abase = Asm[cur];
        const char* bbase = Bsm[cur];
        #pragma unroll
        for (int s = 0; s < 2; ++s) {          // two 32-K WMMA steps per stage
            // A 16x32 frag: lanes 0-15: K0-7 & K16-23; lanes 16-31: K8-15 & K24-31
            v16bf af[2];
            #pragma unroll
            for (int i = 0; i < 2; ++i) {
                const char* pa = abase + (wm * 32 + i * 16 + lrow) * ROWB
                                       + s * 64 + half * 16;
                v8bf lo = *(const v8bf*)pa;
                v8bf hi = *(const v8bf*)(pa + 32);
                af[i] = __builtin_shufflevector(lo, hi,
                    0,1,2,3,4,5,6,7,8,9,10,11,12,13,14,15);
            }
            // B 32x16 frag: lanes 0-15: K0-7 & K16-23 of col n; lanes 16-31: K8-15 & K24-31
            #pragma unroll
            for (int j = 0; j < NJ; ++j) {
                const char* pb = bbase + (wn * (BN_T / 2) + j * 16 + lrow) * ROWB
                                       + s * 64 + half * 16;
                v8bf lo = *(const v8bf*)pb;
                v8bf hi = *(const v8bf*)(pb + 32);
                v16bf bfrag = __builtin_shufflevector(lo, hi,
                    0,1,2,3,4,5,6,7,8,9,10,11,12,13,14,15);
                #pragma unroll
                for (int i = 0; i < 2; ++i)
                    acc[i][j] = __builtin_amdgcn_wmma_f32_16x16x32_bf16(
                        false, af[i], false, bfrag,
                        (short)0, acc[i][j], false, false);
            }
        }
        if (wid == 0 && (kk + 1) < nk) __builtin_amdgcn_s_wait_tensorcnt(0);
        __syncthreads();
    }

    // epilogue: D layout: VGPR r -> row half*8+r, col = lane&15
    float*  Cf = (float*)Cv;
    bf16_t* Cb = (bf16_t*)Cv;
    const long cbase = (long)blockIdx.z * strideC + (long)blockIdx.y * BM * ldc
                     + (long)blockIdx.x * BN_T;
    #pragma unroll
    for (int i = 0; i < 2; ++i) {
        #pragma unroll
        for (int j = 0; j < NJ; ++j) {
            int colL = wn * (BN_T / 2) + j * 16 + lrow;
            float bv = bias ? bias[blockIdx.x * BN_T + colL] : 0.0f;
            #pragma unroll
            for (int r = 0; r < 8; ++r) {
                int row = wm * 32 + i * 16 + half * 8 + r;
                float v = acc[i][j][r] + bv;
                if (ACT == 2) v = gelu_exact(v);
                if (OUT_BF16) Cb[cbase + (long)row * ldc + colL] = (bf16_t)v;
                else          Cf[cbase + (long)row * ldc + colL] = v;
            }
        }
    }
}

// =====================================================================
// Small helper kernels
// =====================================================================

// f32 -> bf16, 8 elems/thread (n % 2048 == 0 for all uses)
__global__ __launch_bounds__(256)
void cvt_kernel(const float* __restrict__ src, bf16_t* __restrict__ dst, size_t n)
{
    size_t i = ((size_t)blockIdx.x * 256 + threadIdx.x) * 8;
    if (i >= n) return;
    float4 a = *(const float4*)(src + i);
    float4 b = *(const float4*)(src + i + 4);
    v8bf o;
    o[0]=(bf16_t)a.x; o[1]=(bf16_t)a.y; o[2]=(bf16_t)a.z; o[3]=(bf16_t)a.w;
    o[4]=(bf16_t)b.x; o[5]=(bf16_t)b.y; o[6]=(bf16_t)b.z; o[7]=(bf16_t)b.w;
    *(v8bf*)(dst + i) = o;
}

// v (B,L,H*DH) bf16 -> vT (B,H,DH,L) bf16
__global__ __launch_bounds__(256)
void vtrans_kernel(const bf16_t* __restrict__ vb, bf16_t* __restrict__ vT)
{
    size_t idx = (size_t)blockIdx.x * 256 + threadIdx.x;   // over B*H*DH*L
    int s = (int)(idx % L_);
    size_t t = idx / L_;
    int d = (int)(t % DH_);
    size_t t2 = t / DH_;
    int h = (int)(t2 % H_);
    int b = (int)(t2 / H_);
    vT[idx] = vb[((size_t)b * L_ + s) * D_ + h * DH_ + d];
}

// mean/std over L for each (b, c)
__global__ void stats_kernel(const float* __restrict__ xe,
                             float* __restrict__ meanb, float* __restrict__ stdb)
{
    int idx = blockIdx.x * blockDim.x + threadIdx.x;
    if (idx >= B_ * CIN_) return;
    int b = idx / CIN_, c = idx % CIN_;
    float s = 0.f, s2 = 0.f;
    for (int l = 0; l < L_; ++l) {
        float v = xe[((size_t)b * L_ + l) * CIN_ + c];
        s += v; s2 += v * v;
    }
    float m = s / L_;
    float var = s2 / L_ - m * m;
    meanb[idx] = m;
    stdb[idx]  = sqrtf(var + 1e-5f);
}

// projector: circ conv over C_IN (L input channels) + concat stats + MLP + head
__global__ __launch_bounds__(128)
void projector_kernel(const float* __restrict__ xe, const float* __restrict__ convw,
                      const float* __restrict__ stats,
                      const float* __restrict__ w0, const float* __restrict__ b0,
                      const float* __restrict__ w1, const float* __restrict__ b1,
                      const float* __restrict__ w2,
                      float* __restrict__ outp, int n_out, int apply_exp)
{
    __shared__ float z[32], h0[PH_], h1[PH_];
    int b = blockIdx.x, t = threadIdx.x;
    if (t < 16) {
        float acc = 0.f;
        int cm = (t + 15) & 15, cp = (t + 1) & 15;
        for (int l = 0; l < L_; ++l) {
            const float* xr = xe + ((size_t)b * L_ + l) * CIN_;
            const float* wr = convw + l * 3;
            acc += wr[0] * xr[cm] + wr[1] * xr[t] + wr[2] * xr[cp];
        }
        z[t] = acc;
    } else if (t < 32) {
        z[t] = stats[b * CIN_ + (t - 16)];
    }
    __syncthreads();
    {
        float a = b0[t];
        for (int j = 0; j < 32; ++j) a += w0[t * 32 + j] * z[j];
        h0[t] = fmaxf(a, 0.f);
    }
    __syncthreads();
    {
        float a = b1[t];
        for (int j = 0; j < PH_; ++j) a += w1[t * PH_ + j] * h0[j];
        h1[t] = fmaxf(a, 0.f);
    }
    __syncthreads();
    for (int o = t; o < n_out; o += 128) {
        float a = 0.f;
        for (int j = 0; j < PH_; ++j) a += w2[o * PH_ + j] * h1[j];
        if (apply_exp) a = __expf(a);
        outp[(size_t)b * n_out + o] = a;
    }
}

// token circular conv (C_IN -> D over length L) + sinusoidal positional embedding
__global__ void embed_kernel(const float* __restrict__ xe,
                             const float* __restrict__ tokw,
                             float* __restrict__ x)
{
    size_t idx = (size_t)blockIdx.x * 256 + threadIdx.x;   // over M_*D_
    int d = (int)(idx % D_);
    size_t ml = idx / D_;
    int l = (int)(ml % L_);
    int b = (int)(ml / L_);
    float acc = 0.f;
    #pragma unroll
    for (int kk = 0; kk < 3; ++kk) {
        int ls = (l + kk - 1 + L_) % L_;
        const float* xr = xe + ((size_t)b * L_ + ls) * CIN_;
        const float* wr = tokw + (size_t)d * CIN_ * 3 + kk;
        #pragma unroll
        for (int c = 0; c < CIN_; ++c) acc += wr[c * 3] * xr[c];
    }
    int j = d >> 1;
    float div = __expf(-(2.0f * j) * (9.210340371976184f / D_));
    float arg = (float)l * div;
    acc += (d & 1) ? cosf(arg) : sinf(arg);
    x[idx] = acc;
}

// softmax over s of scale*(S*tau[b] + delta[b,s]); f32 in, bf16 probs out
__global__ __launch_bounds__(256)
void softmax_kernel(const float* __restrict__ S, bf16_t* __restrict__ P,
                    const float* __restrict__ tau, const float* __restrict__ delta)
{
    __shared__ float red[256];
    int row = blockIdx.x;                 // b*L + l
    int b = row / L_;
    const float* sr = S + (size_t)row * L_;
    const float t = tau[b];
    const float scale = 0.125f;           // 1/sqrt(DH)
    int tid = threadIdx.x;
    float v0 = scale * (sr[tid]       * t + delta[b * L_ + tid]);
    float v1 = scale * (sr[tid + 256] * t + delta[b * L_ + tid + 256]);
    red[tid] = fmaxf(v0, v1); __syncthreads();
    for (int s = 128; s > 0; s >>= 1) {
        if (tid < s) red[tid] = fmaxf(red[tid], red[tid + s]);
        __syncthreads();
    }
    float mx = red[0]; __syncthreads();
    float e0 = __expf(v0 - mx), e1 = __expf(v1 - mx);
    red[tid] = e0 + e1; __syncthreads();
    for (int s = 128; s > 0; s >>= 1) {
        if (tid < s) red[tid] += red[tid + s];
        __syncthreads();
    }
    float inv = 1.0f / red[0];
    bf16_t* pr = P + (size_t)row * L_;
    pr[tid]       = (bf16_t)(e0 * inv);
    pr[tid + 256] = (bf16_t)(e1 * inv);
}

// x = LayerNorm(x + add) * g + b   (in place over x), row-wise over D=512
__global__ __launch_bounds__(256)
void ln_kernel(float* __restrict__ x, const float* __restrict__ add,
               const float* __restrict__ g, const float* __restrict__ be)
{
    __shared__ float red[256];
    int row = blockIdx.x, tid = threadIdx.x;
    float* xr = x + (size_t)row * D_;
    const float* ar = add + (size_t)row * D_;
    float t0 = xr[tid] + ar[tid];
    float t1 = xr[tid + 256] + ar[tid + 256];
    red[tid] = t0 + t1; __syncthreads();
    for (int s = 128; s > 0; s >>= 1) { if (tid < s) red[tid] += red[tid + s]; __syncthreads(); }
    float mean = red[0] / D_; __syncthreads();
    red[tid] = t0 * t0 + t1 * t1; __syncthreads();
    for (int s = 128; s > 0; s >>= 1) { if (tid < s) red[tid] += red[tid + s]; __syncthreads(); }
    float var = red[0] / D_ - mean * mean;
    float inv = rsqrtf(var + 1e-5f);
    xr[tid]       = (t0 - mean) * inv * g[tid]       + be[tid];
    xr[tid + 256] = (t1 - mean) * inv * g[tid + 256] + be[tid + 256];
}

// out = gelu(LayerNorm(x)) * mark[row]
__global__ __launch_bounds__(256)
void fin_kernel(const float* __restrict__ x, const float* __restrict__ mark,
                const float* __restrict__ g, const float* __restrict__ be,
                float* __restrict__ out)
{
    __shared__ float red[256];
    int row = blockIdx.x, tid = threadIdx.x;
    const float* xr = x + (size_t)row * D_;
    float t0 = xr[tid], t1 = xr[tid + 256];
    red[tid] = t0 + t1; __syncthreads();
    for (int s = 128; s > 0; s >>= 1) { if (tid < s) red[tid] += red[tid + s]; __syncthreads(); }
    float mean = red[0] / D_; __syncthreads();
    red[tid] = t0 * t0 + t1 * t1; __syncthreads();
    for (int s = 128; s > 0; s >>= 1) { if (tid < s) red[tid] += red[tid + s]; __syncthreads(); }
    float var = red[0] / D_ - mean * mean;
    float inv = rsqrtf(var + 1e-5f);
    float mk = mark[row];
    float* orow = out + (size_t)row * D_;
    orow[tid]       = gelu_exact((t0 - mean) * inv * g[tid]       + be[tid])       * mk;
    orow[tid + 256] = gelu_exact((t1 - mean) * inv * g[tid + 256] + be[tid + 256]) * mk;
}

// out[b,c] = dot(xa[b,:], proj_w[c,:]) + proj_b[c]  (K = L*D = 262144)
__global__ __launch_bounds__(256)
void proj_kernel(const float* __restrict__ xa, const float* __restrict__ pw,
                 const float* __restrict__ pb, float* __restrict__ out)
{
    __shared__ float red[256];
    int c = blockIdx.x, b = blockIdx.y, tid = threadIdx.x;
    const float* xr = xa + (size_t)b * (L_ * D_);
    const float* wr = pw + (size_t)c * (L_ * D_);
    float acc = 0.f;
    for (int i = tid; i < L_ * D_; i += 256) acc += xr[i] * wr[i];
    red[tid] = acc; __syncthreads();
    for (int s = 128; s > 0; s >>= 1) { if (tid < s) red[tid] += red[tid + s]; __syncthreads(); }
    if (tid == 0) out[b * NC_ + c] = red[0] + pb[c];
}

// =====================================================================
// Host orchestration
// =====================================================================
template<int BN_T, int ACT, int OB>
static inline void launch_gemm(hipStream_t s,
                               const bf16_t* A, int lda, long sA,
                               const bf16_t* Bm, int ldb, long sB,
                               void* C, int ldc, long sC,
                               const float* bias, int M, int N, int K, int batch)
{
    dim3 g(N / BN_T, M / BM, batch);
    gemm_bf16_tdm<BN_T, ACT, OB><<<g, 256, 0, s>>>(A, lda, sA, Bm, ldb, sB,
                                                   C, ldc, sC, bias, K);
}

static inline void launch_cvt(hipStream_t s, const float* src, bf16_t* dst, size_t n)
{
    cvt_kernel<<<(unsigned)(n / 2048), 256, 0, s>>>(src, dst, n);
}

extern "C" void kernel_launch(void* const* d_in, const int* in_sizes, int n_in,
                              void* d_out, int out_size, void* d_ws, size_t ws_size,
                              hipStream_t stream)
{
    (void)in_sizes; (void)n_in; (void)out_size; (void)ws_size;

    const float* x_enc      = (const float*)d_in[0];
    const float* x_mark_enc = (const float*)d_in[1];
    const float* tok_w      = (const float*)d_in[2];
    const float* Wq = (const float*)d_in[3];  const float* bq = (const float*)d_in[4];
    const float* Wk = (const float*)d_in[5];  const float* bk = (const float*)d_in[6];
    const float* Wv = (const float*)d_in[7];  const float* bv = (const float*)d_in[8];
    const float* Wo = (const float*)d_in[9];  const float* bo = (const float*)d_in[10];
    const float* W1 = (const float*)d_in[11]; const float* b1 = (const float*)d_in[12];
    const float* W2 = (const float*)d_in[13]; const float* b2 = (const float*)d_in[14];
    const float* ln1_g = (const float*)d_in[15]; const float* ln1_b = (const float*)d_in[16];
    const float* ln2_g = (const float*)d_in[17]; const float* ln2_b = (const float*)d_in[18];
    const float* norm_g = (const float*)d_in[19]; const float* norm_b = (const float*)d_in[20];
    const float* proj_w = (const float*)d_in[21]; const float* proj_b = (const float*)d_in[22];
    const float* tau_conv_w = (const float*)d_in[23];
    const float* tau_w0 = (const float*)d_in[24]; const float* tau_b0 = (const float*)d_in[25];
    const float* tau_w1 = (const float*)d_in[26]; const float* tau_b1 = (const float*)d_in[27];
    const float* tau_w2 = (const float*)d_in[28];
    const float* del_conv_w = (const float*)d_in[29];
    const float* del_w0 = (const float*)d_in[30]; const float* del_b0 = (const float*)d_in[31];
    const float* del_w1 = (const float*)d_in[32]; const float* del_b1 = (const float*)d_in[33];
    const float* del_w2 = (const float*)d_in[34];
    float* out = (float*)d_out;

    // ---- workspace carve-up (bytes, 256B aligned blocks) ----
    char* base = (char*)d_ws;
    size_t off = 0;
    auto alloc = [&](size_t bytes) -> void* {
        void* p = base + off;
        off += (bytes + 255) & ~(size_t)255;
        return p;
    };
    float*  x      = (float*) alloc((size_t)M_ * D_ * 4);
    float*  tmp    = (float*) alloc((size_t)M_ * D_ * 4);
    float*  scores = (float*) alloc((size_t)B_ * L_ * L_ * 4);
    bf16_t* x_bf   = (bf16_t*)alloc((size_t)M_ * D_ * 2);
    bf16_t* q_bf   = (bf16_t*)alloc((size_t)M_ * D_ * 2);
    bf16_t* k_bf   = (bf16_t*)alloc((size_t)M_ * D_ * 2);
    bf16_t* v_bf   = (bf16_t*)alloc((size_t)M_ * D_ * 2);
    bf16_t* o_bf   = (bf16_t*)alloc((size_t)M_ * D_ * 2);
    bf16_t* vT     = (bf16_t*)alloc((size_t)M_ * D_ * 2);
    bf16_t* h_bf   = (bf16_t*)alloc((size_t)M_ * DFF_ * 2);
    bf16_t* sm_bf  = (bf16_t*)alloc((size_t)B_ * L_ * L_ * 2);
    bf16_t* wq_bf  = (bf16_t*)alloc((size_t)NL_ * D_ * D_ * 2);
    bf16_t* wk_bf  = (bf16_t*)alloc((size_t)NL_ * D_ * D_ * 2);
    bf16_t* wv_bf  = (bf16_t*)alloc((size_t)NL_ * D_ * D_ * 2);
    bf16_t* wo_bf  = (bf16_t*)alloc((size_t)NL_ * D_ * D_ * 2);
    bf16_t* w1_bf  = (bf16_t*)alloc((size_t)NL_ * DFF_ * D_ * 2);
    bf16_t* w2_bf  = (bf16_t*)alloc((size_t)NL_ * D_ * DFF_ * 2);
    float*  meanb  = (float*) alloc(B_ * CIN_ * 4);
    float*  stdb   = (float*) alloc(B_ * CIN_ * 4);
    float*  tau    = (float*) alloc(B_ * 4);
    float*  delta  = (float*) alloc((size_t)B_ * L_ * 4);

    // ---- weights -> bf16 (once) ----
    launch_cvt(stream, Wq, wq_bf, (size_t)NL_ * D_ * D_);
    launch_cvt(stream, Wk, wk_bf, (size_t)NL_ * D_ * D_);
    launch_cvt(stream, Wv, wv_bf, (size_t)NL_ * D_ * D_);
    launch_cvt(stream, Wo, wo_bf, (size_t)NL_ * D_ * D_);
    launch_cvt(stream, W1, w1_bf, (size_t)NL_ * DFF_ * D_);
    launch_cvt(stream, W2, w2_bf, (size_t)NL_ * D_ * DFF_);

    // ---- stats, tau/delta projectors ----
    stats_kernel<<<(B_ * CIN_ + 63) / 64, 64, 0, stream>>>(x_enc, meanb, stdb);
    projector_kernel<<<B_, 128, 0, stream>>>(x_enc, tau_conv_w, stdb,
        tau_w0, tau_b0, tau_w1, tau_b1, tau_w2, tau, 1, 1);
    projector_kernel<<<B_, 128, 0, stream>>>(x_enc, del_conv_w, meanb,
        del_w0, del_b0, del_w1, del_b1, del_w2, delta, L_, 0);

    // ---- embedding ----
    embed_kernel<<<((size_t)M_ * D_) / 256, 256, 0, stream>>>(x_enc, tok_w, x);
    launch_cvt(stream, x, x_bf, (size_t)M_ * D_);

    // ---- transformer layers ----
    for (int i = 0; i < NL_; ++i) {
        const bf16_t* Wqi = wq_bf + (size_t)i * D_ * D_;
        const bf16_t* Wki = wk_bf + (size_t)i * D_ * D_;
        const bf16_t* Wvi = wv_bf + (size_t)i * D_ * D_;
        const bf16_t* Woi = wo_bf + (size_t)i * D_ * D_;
        const bf16_t* W1i = w1_bf + (size_t)i * DFF_ * D_;
        const bf16_t* W2i = w2_bf + (size_t)i * D_ * DFF_;

        // Q, K, V projections (bf16 out)
        launch_gemm<128, 0, 1>(stream, x_bf, D_, 0, Wqi, D_, 0, q_bf, D_, 0,
                               bq + i * D_, M_, D_, D_, 1);
        launch_gemm<128, 0, 1>(stream, x_bf, D_, 0, Wki, D_, 0, k_bf, D_, 0,
                               bk + i * D_, M_, D_, D_, 1);
        launch_gemm<128, 0, 1>(stream, x_bf, D_, 0, Wvi, D_, 0, v_bf, D_, 0,
                               bv + i * D_, M_, D_, D_, 1);
        vtrans_kernel<<<((size_t)M_ * D_) / 256, 256, 0, stream>>>(v_bf, vT);

        // attention, per head
        for (int h = 0; h < H_; ++h) {
            // scores[b] = q_h @ k_h^T (512x512, K=64), f32 out
            launch_gemm<128, 0, 0>(stream,
                q_bf + h * DH_, D_, (long)L_ * D_,
                k_bf + h * DH_, D_, (long)L_ * D_,
                scores, L_, (long)L_ * L_,
                nullptr, L_, L_, DH_, B_);
            softmax_kernel<<<B_ * L_, 256, 0, stream>>>(scores, sm_bf, tau, delta);
            // o_h[b] = P @ v_h via vT (N=DH rows of length K=L), bf16 out
            launch_gemm<64, 0, 1>(stream,
                sm_bf, L_, (long)L_ * L_,
                vT + (size_t)h * DH_ * L_, L_, (long)D_ * L_,
                o_bf + h * DH_, D_, (long)L_ * D_,
                nullptr, L_, DH_, L_, B_);
        }

        // output projection (f32 out), residual + LN1
        launch_gemm<128, 0, 0>(stream, o_bf, D_, 0, Woi, D_, 0, tmp, D_, 0,
                               bo + i * D_, M_, D_, D_, 1);
        ln_kernel<<<M_, 256, 0, stream>>>(x, tmp, ln1_g + i * D_, ln1_b + i * D_);
        launch_cvt(stream, x, x_bf, (size_t)M_ * D_);

        // FFN
        launch_gemm<128, 2, 1>(stream, x_bf, D_, 0, W1i, D_, 0, h_bf, DFF_, 0,
                               b1 + i * DFF_, M_, DFF_, D_, 1);
        launch_gemm<128, 0, 0>(stream, h_bf, DFF_, 0, W2i, DFF_, 0, tmp, D_, 0,
                               b2 + i * D_, M_, D_, DFF_, 1);
        ln_kernel<<<M_, 256, 0, stream>>>(x, tmp, ln2_g + i * D_, ln2_b + i * D_);
        launch_cvt(stream, x, x_bf, (size_t)M_ * D_);
    }

    // ---- final LN + gelu + mask -> tmp, then projection ----
    fin_kernel<<<M_, 256, 0, stream>>>(x, x_mark_enc, norm_g, norm_b, tmp);
    {
        dim3 g(NC_, B_, 1);
        proj_kernel<<<g, 256, 0, stream>>>(tmp, proj_w, proj_b, out);
    }
}
